// PerceiverAttention_1726576857551
// MI455X (gfx1250) — matrix operand order repaired and checked by reading, not compiled, not run on hardware
//
#include <hip/hip_runtime.h>
#include <hip/hip_bf16.h>
#include <stdint.h>

typedef __bf16 bf16;
typedef __attribute__((ext_vector_type(16))) __bf16 v16bf;
typedef __attribute__((ext_vector_type(8)))  float  v8f;

#define NT     64      // b*T tiles
#define N1     2048
#define N2     64
#define DIMD   1024
#define INNER  512
#define HEADS  8
#define DH     64
#define NKV    2112    // N1 + N2
#define NXROWS (NT * N1)   // 131072
#define NLROWS (NT * N2)   // 4096
#define NROWS  (NXROWS + NLROWS)

// ---------------- fragment helpers (wave32, 16x16x32 bf16 WMMA) ----------------

__device__ __forceinline__ v8f zero8() {
  v8f z;
#pragma unroll
  for (int i = 0; i < 8; ++i) z[i] = 0.0f;
  return z;
}

// A fragment: 16 rows (M) x 32 k, src row-major with leading dim ld.
// lane = row%16 in each half; K order per ISA: V0..3 -> {0..7}+hi*8, V4..7 -> {16..23}+hi*8
__device__ __forceinline__ v16bf load_a_frag(const bf16* __restrict__ p, int row0, int k0, int ld) {
  int lane = threadIdx.x & 31;
  int r = lane & 15, hi = lane >> 4;
  const bf16* rp = p + (size_t)(row0 + r) * ld + k0;
  v16bf f;
#pragma unroll
  for (int v = 0; v < 8; ++v) {
    int kk = ((v & 4) << 2) + hi * 8 + (v & 3) * 2;
    f[2 * v]     = rp[kk];
    f[2 * v + 1] = rp[kk + 1];
  }
  return f;
}

// B fragment: 32 k x 16 cols (N). Source is stored N-major (row = n, ld over k).
// K order per ISA (same as A): V0..3 -> {0..7}+hi*8, V4..7 -> {16..23}+hi*8
__device__ __forceinline__ v16bf load_b_frag(const bf16* __restrict__ p, int n0, int k0, int ld) {
  int lane = threadIdx.x & 31;
  int n = lane & 15, hi = lane >> 4;
  const bf16* rp = p + (size_t)(n0 + n) * ld + k0;
  v16bf f;
#pragma unroll
  for (int v = 0; v < 8; ++v) {
    int kk = ((v & 4) << 2) + hi * 8 + (v & 3) * 2;
    f[2 * v]     = rp[kk];
    f[2 * v + 1] = rp[kk + 1];
  }
  return f;
}

__device__ __forceinline__ v8f wmma_bf16(v16bf a, v16bf b, v8f c) {
  return __builtin_amdgcn_wmma_f32_16x16x32_bf16(false, a, false, b, (short)0, c, false, false);
}

// ---------------- kernel 1: LN row statistics ----------------

__global__ void __launch_bounds__(256) stats_kernel(const float* __restrict__ x,
                                                    const float* __restrict__ lat,
                                                    float* __restrict__ mu,
                                                    float* __restrict__ rstd) {
  int row = blockIdx.x;
  const float* src = (row < NXROWS) ? (x + (size_t)row * DIMD)
                                    : (lat + (size_t)(row - NXROWS) * DIMD);
  int tid = threadIdx.x;
  float4 v = ((const float4*)src)[tid];
  float s = v.x + v.y + v.z + v.w;
  float q = v.x * v.x + v.y * v.y + v.z * v.z + v.w * v.w;
  __shared__ float ss[256], sq[256];
  ss[tid] = s; sq[tid] = q;
  __syncthreads();
  for (int st = 128; st > 0; st >>= 1) {
    if (tid < st) { ss[tid] += ss[tid + st]; sq[tid] += sq[tid + st]; }
    __syncthreads();
  }
  if (tid == 0) {
    float m = ss[0] * (1.0f / DIMD);
    float var = sq[0] * (1.0f / DIMD) - m * m;
    mu[row] = m;
    rstd[row] = rsqrtf(var + 1e-5f);
  }
}

// ---------------- kernel 2: transpose + bf16 convert weights ----------------

__global__ void __launch_bounds__(256) prep_weights(const float* __restrict__ Wq,
                                                    const float* __restrict__ Wkv,
                                                    const float* __restrict__ Wout,
                                                    bf16* __restrict__ WqT,
                                                    bf16* __restrict__ WkvT,
                                                    bf16* __restrict__ WoutT) {
  int idx = blockIdx.x * 256 + threadIdx.x;
  const int SZQ = INNER * DIMD;       // 524288
  const int SZKV = 2 * INNER * DIMD;  // 1048576
  if (idx < SZQ) {                            // WqT: INNER rows x DIMD k
    int n = idx >> 10, k = idx & 1023;
    WqT[idx] = (bf16)Wq[(size_t)k * INNER + n];
  } else if (idx < SZQ + SZKV) {              // WkvT: 2*INNER rows x DIMD k
    int i = idx - SZQ;
    int n = i >> 10, k = i & 1023;
    WkvT[i] = (bf16)Wkv[(size_t)k * (2 * INNER) + n];
  } else {                                    // WoutT: DIMD rows x INNER k
    int i = idx - SZQ - SZKV;
    int n = i >> 9, k = i & 511;
    WoutT[i] = (bf16)Wout[(size_t)k * DIMD + n];
  }
}

// ---------------- kernel 3: q = LN(latents) @ Wq (scaled), per-head bf16 ----------------

__global__ void __launch_bounds__(256) q_gemm(const float* __restrict__ lat,
                                              const float* __restrict__ glw,
                                              const float* __restrict__ glb,
                                              const float* __restrict__ mu,
                                              const float* __restrict__ rstd,
                                              const bf16* __restrict__ WqT,
                                              bf16* __restrict__ Qws) {
  int tile = blockIdx.x;
  int tid = threadIdx.x;
  int w = tid >> 5, lane = tid & 31, hi = lane >> 4, ln16 = lane & 15;
  int c0 = w * 64;                      // wave covers 64 cols, 64 rows
  __shared__ bf16 As[64 * 32];

  v8f acc[4][4];
#pragma unroll
  for (int rt = 0; rt < 4; ++rt)
#pragma unroll
    for (int ct = 0; ct < 4; ++ct) acc[rt][ct] = zero8();

  int e = tid * 8, fr = e >> 5, fk = e & 31;
  int lr = tile * N2 + fr;
  const float* src = lat + (size_t)lr * DIMD;
  float fm = mu[NXROWS + lr], frs = rstd[NXROWS + lr];

  for (int ks = 0; ks < DIMD / 32; ++ks) {
    int k0 = ks * 32;
#pragma unroll
    for (int i = 0; i < 8; ++i) {
      int k = k0 + fk + i;
      As[e + i] = (bf16)((src[k] - fm) * frs * glw[k] + glb[k]);
    }
    __syncthreads();
    v16bf af[4];
#pragma unroll
    for (int rt = 0; rt < 4; ++rt) af[rt] = load_a_frag(As, rt * 16, 0, 32);
#pragma unroll
    for (int ct = 0; ct < 4; ++ct) {
      v16bf bf = load_b_frag(WqT, c0 + ct * 16, k0, DIMD);
#pragma unroll
      for (int rt = 0; rt < 4; ++rt) acc[rt][ct] = wmma_bf16(af[rt], bf, acc[rt][ct]);
    }
    __syncthreads();
  }

  const float scale = 0.125f;  // DIM_HEAD^-0.5
#pragma unroll
  for (int rt = 0; rt < 4; ++rt)
#pragma unroll
    for (int ct = 0; ct < 4; ++ct) {
      int n = c0 + ct * 16 + ln16;
      int h = n >> 6, d = n & 63;
#pragma unroll
      for (int v = 0; v < 8; ++v) {
        int m = rt * 16 + v + 8 * hi;
        Qws[((size_t)(tile * HEADS + h) * N2 + m) * DH + d] = (bf16)(acc[rt][ct][v] * scale);
      }
    }
}

// ---------------- kernel 4: [xn;ln] @ Wkv -> K (j,d) and V^T (d,j), bf16 ----------------

__global__ void __launch_bounds__(256) kv_gemm(const float* __restrict__ x,
                                               const float* __restrict__ lat,
                                               const float* __restrict__ gmw,
                                               const float* __restrict__ gmb,
                                               const float* __restrict__ glw,
                                               const float* __restrict__ glb,
                                               const float* __restrict__ mu,
                                               const float* __restrict__ rstd,
                                               const bf16* __restrict__ WkvT,
                                               bf16* __restrict__ Kws,
                                               bf16* __restrict__ Vtws) {
  int tile = blockIdx.x / 66;
  int jbase = (blockIdx.x % 66) * 32;   // 66*32 = 2112
  int tid = threadIdx.x;
  int w = tid >> 5, lane = tid & 31, hi = lane >> 4, ln16 = lane & 15;
  int c0 = w * 128;                     // wave covers 128 cols, 32 rows
  __shared__ bf16 As[32 * 32];

  v8f acc[2][8];
#pragma unroll
  for (int rt = 0; rt < 2; ++rt)
#pragma unroll
    for (int ct = 0; ct < 8; ++ct) acc[rt][ct] = zero8();

  // LDS fill assignment: 4 contiguous k per thread, fixed row
  int e = tid * 4, fr = e >> 5, fk = e & 31;
  int j = jbase + fr;
  const float* src; const float* g; const float* bb; float fm, frs;
  if (j < N1) {
    int xr = tile * N1 + j;
    src = x + (size_t)xr * DIMD; g = gmw; bb = gmb; fm = mu[xr]; frs = rstd[xr];
  } else {
    int lr = tile * N2 + (j - N1);
    src = lat + (size_t)lr * DIMD; g = glw; bb = glb;
    fm = mu[NXROWS + lr]; frs = rstd[NXROWS + lr];
  }

  for (int ks = 0; ks < DIMD / 32; ++ks) {
    int k0 = ks * 32;
#pragma unroll
    for (int i = 0; i < 4; ++i) {
      int k = k0 + fk + i;
      As[e + i] = (bf16)((src[k] - fm) * frs * g[k] + bb[k]);
    }
    __syncthreads();
    v16bf af0 = load_a_frag(As, 0, 0, 32);
    v16bf af1 = load_a_frag(As, 16, 0, 32);
#pragma unroll
    for (int ct = 0; ct < 8; ++ct) {
      v16bf bf = load_b_frag(WkvT, c0 + ct * 16, k0, DIMD);
      acc[0][ct] = wmma_bf16(af0, bf, acc[0][ct]);
      acc[1][ct] = wmma_bf16(af1, bf, acc[1][ct]);
    }
    __syncthreads();
  }

#pragma unroll
  for (int rt = 0; rt < 2; ++rt)
#pragma unroll
    for (int ct = 0; ct < 8; ++ct) {
      int n = c0 + ct * 16 + ln16;
#pragma unroll
      for (int v = 0; v < 8; ++v) {
        int jj = jbase + rt * 16 + v + 8 * hi;
        float val = acc[rt][ct][v];
        if (n < INNER) {           // K: (tile,h) x j x d  (row-major over d)
          int h = n >> 6, d = n & 63;
          Kws[((size_t)(tile * HEADS + h) * NKV + jj) * DH + d] = (bf16)val;
        } else {                   // V^T: (tile,h) x d x j (row-major over j)
          int nv = n - INNER; int h = nv >> 6, d = nv & 63;
          Vtws[((size_t)(tile * HEADS + h) * DH + d) * NKV + jj] = (bf16)val;
        }
      }
    }
}

// ---------------- kernel 5: flash attention per (tile, head) ----------------

__global__ void __launch_bounds__(128) attn_kernel(const bf16* __restrict__ Qws,
                                                   const bf16* __restrict__ Kws,
                                                   const bf16* __restrict__ Vtws,
                                                   bf16* __restrict__ AOws) {
  int th = blockIdx.x;                  // tile*HEADS + h
  int tile = th >> 3, h = th & 7;
  int tid = threadIdx.x;
  int w = tid >> 5, lane = tid & 31, hi = lane >> 4, ln16 = lane & 15;
  int m0 = w * 16;                      // 4 waves x 16 query rows = 64
  const bf16* Qh  = Qws  + (size_t)th * N2 * DH;
  const bf16* Kh  = Kws  + (size_t)th * NKV * DH;
  const bf16* Vth = Vtws + (size_t)th * DH * NKV;
  __shared__ bf16 Ps[4][16 * 64];       // per-wave P staging (A-layout transpose)

  v16bf aq0 = load_a_frag(Qh, m0, 0, DH);
  v16bf aq1 = load_a_frag(Qh, m0, 32, DH);

  float mrow[8], lrow[8];
  v8f o[4];
#pragma unroll
  for (int v = 0; v < 8; ++v) { mrow[v] = -1e30f; lrow[v] = 0.0f; }
#pragma unroll
  for (int ot = 0; ot < 4; ++ot) o[ot] = zero8();

  for (int jb = 0; jb < NKV / 64; ++jb) {
    int j0 = jb * 64;
    v8f s[4];
#pragma unroll
    for (int ct = 0; ct < 4; ++ct) {
      s[ct] = zero8();
      v16bf b0 = load_b_frag(Kh, j0 + ct * 16, 0, DH);
      s[ct] = wmma_bf16(aq0, b0, s[ct]);
      v16bf b1 = load_b_frag(Kh, j0 + ct * 16, 32, DH);
      s[ct] = wmma_bf16(aq1, b1, s[ct]);
    }
    // online softmax (rows live in half-waves; reduce across 16 lanes)
#pragma unroll
    for (int v = 0; v < 8; ++v) {
      float t = fmaxf(fmaxf(s[0][v], s[1][v]), fmaxf(s[2][v], s[3][v]));
      t = fmaxf(t, __shfl_xor(t, 1, 32));
      t = fmaxf(t, __shfl_xor(t, 2, 32));
      t = fmaxf(t, __shfl_xor(t, 4, 32));
      t = fmaxf(t, __shfl_xor(t, 8, 32));
      float mn = fmaxf(mrow[v], t);
      float corr = __expf(mrow[v] - mn);
      float rs = 0.0f;
#pragma unroll
      for (int ct = 0; ct < 4; ++ct) { s[ct][v] = __expf(s[ct][v] - mn); rs += s[ct][v]; }
      rs += __shfl_xor(rs, 1, 32);
      rs += __shfl_xor(rs, 2, 32);
      rs += __shfl_xor(rs, 4, 32);
      rs += __shfl_xor(rs, 8, 32);
      lrow[v] = lrow[v] * corr + rs;
      mrow[v] = mn;
#pragma unroll
      for (int ot = 0; ot < 4; ++ot) o[ot][v] *= corr;
    }
    // transpose P (C-layout) -> LDS row-major -> A-fragment layout
#pragma unroll
    for (int ct = 0; ct < 4; ++ct)
#pragma unroll
      for (int v = 0; v < 8; ++v)
        Ps[w][(v + 8 * hi) * 64 + ct * 16 + ln16] = (bf16)s[ct][v];
    __syncthreads();
    v16bf pa0 = load_a_frag(Ps[w], 0, 0, 64);
    v16bf pa1 = load_a_frag(Ps[w], 0, 32, 64);
#pragma unroll
    for (int ot = 0; ot < 4; ++ot) {
      v16bf b0 = load_b_frag(Vth, ot * 16, j0, NKV);
      o[ot] = wmma_bf16(pa0, b0, o[ot]);
      v16bf b1 = load_b_frag(Vth, ot * 16, j0 + 32, NKV);
      o[ot] = wmma_bf16(pa1, b1, o[ot]);
    }
    __syncthreads();
  }
  // normalize and store (tile, n2, INNER) bf16
#pragma unroll
  for (int v = 0; v < 8; ++v) {
    float inv = 1.0f / lrow[v];
    int m = m0 + v + 8 * hi;
#pragma unroll
    for (int ot = 0; ot < 4; ++ot)
      AOws[((size_t)tile * N2 + m) * INNER + h * DH + ot * 16 + ln16] = (bf16)(o[ot][v] * inv);
  }
}

// ---------------- kernel 6: attn_out @ Wout -> f32 output ----------------

__global__ void __launch_bounds__(256) out_gemm(const bf16* __restrict__ AOws,
                                                const bf16* __restrict__ WoutT,
                                                float* __restrict__ out) {
  int tile = blockIdx.x >> 1;
  int half = blockIdx.x & 1;
  int tid = threadIdx.x;
  int w = tid >> 5, lane = tid & 31, hi = lane >> 4, ln16 = lane & 15;
  int c0 = half * 512 + w * 64;
  __shared__ bf16 As[64 * 32];

  v8f acc[4][4];
#pragma unroll
  for (int rt = 0; rt < 4; ++rt)
#pragma unroll
    for (int ct = 0; ct < 4; ++ct) acc[rt][ct] = zero8();

  int e = tid * 8, fr = e >> 5, fk = e & 31;
  const bf16* src = AOws + (size_t)tile * N2 * INNER + (size_t)fr * INNER;

  for (int ks = 0; ks < INNER / 32; ++ks) {
    int k0 = ks * 32;
#pragma unroll
    for (int i = 0; i < 8; ++i) As[e + i] = src[k0 + fk + i];
    __syncthreads();
    v16bf af[4];
#pragma unroll
    for (int rt = 0; rt < 4; ++rt) af[rt] = load_a_frag(As, rt * 16, 0, 32);
#pragma unroll
    for (int ct = 0; ct < 4; ++ct) {
      v16bf bf = load_b_frag(WoutT, c0 + ct * 16, k0, INNER);
#pragma unroll
      for (int rt = 0; rt < 4; ++rt) acc[rt][ct] = wmma_bf16(af[rt], bf, acc[rt][ct]);
    }
    __syncthreads();
  }

#pragma unroll
  for (int rt = 0; rt < 4; ++rt)
#pragma unroll
    for (int ct = 0; ct < 4; ++ct) {
      int n = c0 + ct * 16 + ln16;
#pragma unroll
      for (int v = 0; v < 8; ++v) {
        int m = rt * 16 + v + 8 * hi;
        out[(size_t)tile * N2 * DIMD + (size_t)m * DIMD + n] = acc[rt][ct][v];
      }
    }
}

// ---------------- launcher ----------------

extern "C" void kernel_launch(void* const* d_in, const int* in_sizes, int n_in,
                              void* d_out, int out_size, void* d_ws, size_t ws_size,
                              hipStream_t stream) {
  const float* x      = (const float*)d_in[0];
  const float* lat    = (const float*)d_in[1];
  const float* ln_m_w = (const float*)d_in[2];
  const float* ln_m_b = (const float*)d_in[3];
  const float* ln_l_w = (const float*)d_in[4];
  const float* ln_l_b = (const float*)d_in[5];
  const float* Wq     = (const float*)d_in[6];
  const float* Wkv    = (const float*)d_in[7];
  const float* Wout   = (const float*)d_in[8];
  float* out = (float*)d_out;

  char* ws = (char*)d_ws;
  float* mu   = (float*)ws;
  float* rstd = mu + NROWS;
  size_t off = (size_t)NROWS * 2 * sizeof(float);
  bf16* WqT   = (bf16*)(ws + off); off += (size_t)INNER * DIMD * 2;
  bf16* WkvT  = (bf16*)(ws + off); off += (size_t)2 * INNER * DIMD * 2;
  bf16* WoutT = (bf16*)(ws + off); off += (size_t)INNER * DIMD * 2;
  bf16* Qws   = (bf16*)(ws + off); off += (size_t)NT * HEADS * N2 * DH * 2;
  bf16* Kws   = (bf16*)(ws + off); off += (size_t)NT * HEADS * NKV * DH * 2;
  bf16* Vtws  = (bf16*)(ws + off); off += (size_t)NT * HEADS * DH * NKV * 2;
  bf16* AOws  = (bf16*)(ws + off);

  stats_kernel<<<NROWS, 256, 0, stream>>>(x, lat, mu, rstd);
  prep_weights<<<(4 * INNER * DIMD) / 256, 256, 0, stream>>>(Wq, Wkv, Wout, WqT, WkvT, WoutT);
  q_gemm<<<NT, 256, 0, stream>>>(lat, ln_l_w, ln_l_b, mu, rstd, WqT, Qws);
  kv_gemm<<<NT * 66, 256, 0, stream>>>(x, lat, ln_m_w, ln_m_b, ln_l_w, ln_l_b,
                                       mu, rstd, WkvT, Kws, Vtws);
  attn_kernel<<<NT * HEADS, 128, 0, stream>>>(Qws, Kws, Vtws, AOws);
  out_gemm<<<NT * 2, 256, 0, stream>>>(AOws, WoutT, out);
}
